// FRTBlock_78125455114776
// MI455X (gfx1250) — matrix operand order repaired and checked by reading, not compiled, not run on hardware
//
#include <hip/hip_runtime.h>

// MI455X / gfx1250 implementation: all matmuls on v_wmma_f32_16x16x32_bf16.
// GEMM: per-wave 32x16 output tile, B tiles double-buffered in LDS via
// global_load_async_to_lds_b128 with pipelined s_wait_asynccnt (0x2 steady
// state), A streamed from global with global_prefetch; conv uses an LDS
// im2col tile; row reductions use wave32 shuffles.

typedef __bf16 bf16_t;
typedef __attribute__((ext_vector_type(16))) __bf16 v16bf;
typedef __attribute__((ext_vector_type(8)))  __bf16 v8bf;
typedef __attribute__((ext_vector_type(8)))  float  v8f;

static __device__ __forceinline__ v8f wmma_bf16(v16bf a, v16bf b, v8f c) {
    // 8 args: (neg_a, A, neg_b, B, c_mod, C, reuse_a, reuse_b)
    return __builtin_amdgcn_wmma_f32_16x16x32_bf16(false, a, false, b, (short)0, c, false, false);
}

union AFrag { v16bf v; struct { v8bf lo, hi; } p; };

// ---------------------------------------------------------------------------
// Generic GEMM: out[M][Nout] = A[M][K] (bf16, row-major) x W[Nout][K] (bf16).
// One wave per 32x16 tile (2 WMMAs/k-step sharing one B tile); 8 waves/block
// cover 128 output columns. B tiles double-buffered in LDS via async copies.
// EP: 0 none | 1 affine | 2 affine+relu | 3 bias | 4 bias+gelu | 5 affine+relu+2*res
// TMODE: 0 none | 1 vT-style [(batch*128+col)*4096+s_local] | 2 per-z [col][M]
// ---------------------------------------------------------------------------
template<int EP, int TMODE>
__global__ __launch_bounds__(256) void gemm_bf16_wmma(
    const bf16_t* __restrict__ A, long long aBatch,
    const bf16_t* __restrict__ W, long long bBatch,
    int M, int Kd, int ldOut,
    const float* __restrict__ svec, const float* __restrict__ ovec,
    const float* __restrict__ res,
    float* __restrict__ outF,
    bf16_t* __restrict__ outBF, long long oBatch,
    bf16_t* __restrict__ outT)
{
    __shared__ __attribute__((aligned(64))) bf16_t shB[8][2][16][32]; // per-wave double buffer
    const int tid  = threadIdx.x;
    const int wave = tid >> 5;
    const int lane = tid & 31;
    const int g    = lane >> 4;
    const int ln16 = lane & 15;
    const int z    = blockIdx.z;
    const int m0   = blockIdx.x * 32;
    const int n0   = (blockIdx.y * 8 + wave) * 16;

    const bf16_t* Ab = A + (long long)z * aBatch;
    const bf16_t* Wb = W + (long long)z * bBatch;

    // async staging: lane covers 16 bf16 (32B) of one B-tile row (2 lanes/col)
    const int bcol = lane >> 1;
    const int bko  = (lane & 1) * 16;
    const unsigned ldsBase =
        (unsigned)(unsigned long long)(const void*)&shB[wave][0][0][0]
        + (unsigned)(bcol * 64 + bko * 2);
    const bf16_t* aRow0 = Ab + (long long)(m0 + ln16) * Kd;
    const bf16_t* aRow1 = aRow0 + (long long)16 * Kd;
    const bf16_t* bRowP = Wb + (long long)(n0 + bcol) * Kd + bko;

    v8f acc0 = {}, acc1 = {};

    // prologue: stage first B tile into buffer 0
    {
        const bf16_t* gsrc = bRowP;
        asm volatile("global_load_async_to_lds_b128 %0, %1, off"
                     :: "v"(ldsBase), "v"(gsrc) : "memory");
        asm volatile("global_load_async_to_lds_b128 %0, %1, off offset:16"
                     :: "v"(ldsBase), "v"(gsrc) : "memory");
    }

    int ib = 0;
    for (int kb = 0; kb < Kd; kb += 32, ib ^= 1) {
        const bool hasNext = (kb + 32) < Kd;
        if (hasNext) {  // stage next B tile into the other buffer
            const bf16_t* gsrc = bRowP + kb + 32;
            const unsigned nextAddr = ldsBase + (unsigned)((ib ^ 1) * 1024);
            asm volatile("global_load_async_to_lds_b128 %0, %1, off"
                         :: "v"(nextAddr), "v"(gsrc) : "memory");
            asm volatile("global_load_async_to_lds_b128 %0, %1, off offset:16"
                         :: "v"(nextAddr), "v"(gsrc) : "memory");
        }
        AFrag a0, a1;   // A fragments, contiguous K order (k = 16*g + s), matching B
        a0.p.lo = *(const v8bf*)(aRow0 + kb + 16 * g);
        a0.p.hi = *(const v8bf*)(aRow0 + kb + 16 * g + 8);
        a1.p.lo = *(const v8bf*)(aRow1 + kb + 16 * g);
        a1.p.hi = *(const v8bf*)(aRow1 + kb + 16 * g + 8);
        __builtin_prefetch(aRow0 + kb + 64, 0, 1);
        __builtin_prefetch(aRow1 + kb + 64, 0, 1);
        if (hasNext) asm volatile("s_wait_asynccnt 0x2" ::: "memory"); // current done, next in flight
        else         asm volatile("s_wait_asynccnt 0x0" ::: "memory");
        v16bf b = *(const v16bf*)&shB[wave][ib][ln16][16 * g];
        acc0 = wmma_bf16(a0.v, b, acc0);
        acc1 = wmma_bf16(a1.v, b, acc1);
    }

    #pragma unroll
    for (int t = 0; t < 2; ++t) {
        const v8f acc = t ? acc1 : acc0;
        #pragma unroll
        for (int r = 0; r < 8; ++r) {
            const int mrow = m0 + t * 16 + r + 8 * g; // C/D layout: VGPR r -> M = r + 8*(lane>=16)
            const int col  = n0 + ln16;
            float val = acc[r];
            if (EP == 1 || EP == 2 || EP == 5) val = val * svec[col] + ovec[col];
            if (EP == 2 || EP == 5) val = fmaxf(val, 0.0f);
            if (EP == 3 || EP == 4) val += svec[col];
            if (EP == 4) { // tanh-approx gelu (jax default)
                float tt = val;
                val = 0.5f * tt * (1.0f + tanhf(0.7978845608028654f * (tt + 0.044715f * tt * tt * tt)));
            }
            if (EP == 5) val += 2.0f * res[(long long)mrow * ldOut + col];
            if (outF)  outF[(long long)mrow * ldOut + col] = val;
            if (outBF) outBF[(long long)z * oBatch + (long long)mrow * ldOut + col] = (bf16_t)val;
            if (TMODE == 1) {
                const int nb = mrow >> 12, sl = mrow & 4095;
                outT[(((long long)nb * 128) + col) * 4096 + sl] = (bf16_t)val;
            } else if (TMODE == 2) {
                outT[(((long long)z * 128) + col) * (long long)M + mrow] = (bf16_t)val;
            }
        }
    }
}

// ---------------------------------------------------------------------------
// Conv3x3 stride-2 pad-1 (Cin=64 -> C=128) + bias + LayerNorm(C), fused.
// im2col tile (16 positions x 576) built in LDS; WMMA over K=576.
// ---------------------------------------------------------------------------
__global__ __launch_bounds__(256) void conv_ln_kernel(
    const float* __restrict__ x, const bf16_t* __restrict__ Wc,
    const float* __restrict__ bias,
    const float* __restrict__ lng, const float* __restrict__ lnb,
    float* __restrict__ xcF, bf16_t* __restrict__ xcBF)
{
    __shared__ __attribute__((aligned(64))) bf16_t shA[16][576];
    __shared__ float shO[16][128];
    __shared__ float shMean[16], shRstd[16];

    const int tid = threadIdx.x;
    const int m0  = blockIdx.x * 16;         // 16 consecutive positions, same row
    const int nImg = m0 >> 12;
    const int oy   = (m0 >> 6) & 63;
    const int ox0  = m0 & 63;

    for (int e = tid; e < 16 * 576; e += 256) {
        const int p = e / 576, kidx = e - p * 576;
        const int ci = kidx / 9, r9 = kidx - ci * 9;
        const int ky = r9 / 3, kx = r9 - ky * 3;
        const int iy = oy * 2 - 1 + ky;
        const int ix = (ox0 + p) * 2 - 1 + kx;
        float v = 0.0f;
        if (iy >= 0 && iy < 128 && ix >= 0 && ix < 128)
            v = x[(((long long)nImg * 64 + ci) * 128 + iy) * 128 + ix];
        shA[p][kidx] = (bf16_t)v;
    }
    __syncthreads();

    const int wave = tid >> 5, lane = tid & 31, g = lane >> 4, ln16 = lane & 15;
    const int n0 = wave * 16;
    v8f acc = {};
    for (int kb = 0; kb < 576; kb += 32) {
        AFrag a;
        a.p.lo = *(const v8bf*)&shA[ln16][kb + 16 * g];
        a.p.hi = *(const v8bf*)&shA[ln16][kb + 16 * g + 8];
        v16bf b = *(const v16bf*)(Wc + (long long)(n0 + ln16) * 576 + kb + 16 * g);
        acc = wmma_bf16(a.v, b, acc);
    }
    #pragma unroll
    for (int r = 0; r < 8; ++r)
        shO[r + 8 * g][n0 + ln16] = acc[r] + bias[n0 + ln16];
    __syncthreads();

    if (tid < 16) {
        float s = 0.0f, s2 = 0.0f;
        for (int c = 0; c < 128; ++c) { float v = shO[tid][c]; s += v; s2 += v * v; }
        const float m = s * (1.0f / 128.0f);
        shMean[tid] = m;
        shRstd[tid] = rsqrtf(s2 * (1.0f / 128.0f) - m * m + 1e-5f);
    }
    __syncthreads();

    const int p = tid >> 4, c0 = (tid & 15) * 8;
    const float m = shMean[p], rs = shRstd[p];
    for (int c = c0; c < c0 + 8; ++c) {
        const float v = (shO[p][c] - m) * rs * lng[c] + lnb[c];
        const long long idx = (long long)(m0 + p) * 128 + c;
        xcF[idx] = v;
        xcBF[idx] = (bf16_t)v;
    }
}

// Per-position L2 norm over C=128 (one position per wave). TRANS=1 writes [C][HW].
template<int TRANS>
__global__ __launch_bounds__(256) void l2norm_kernel(
    const float* __restrict__ in, bf16_t* __restrict__ out)
{
    const int wave = threadIdx.x >> 5, lane = threadIdx.x & 31;
    const long long s = (long long)blockIdx.x * 8 + wave;
    const float4 v = *(const float4*)(in + s * 128 + lane * 4);
    float vv[4] = { v.x, v.y, v.z, v.w };
    float ss = vv[0]*vv[0] + vv[1]*vv[1] + vv[2]*vv[2] + vv[3]*vv[3];
    for (int o = 16; o >= 1; o >>= 1) ss += __shfl_xor(ss, o, 32);
    const float sc = 1.0f / fmaxf(sqrtf(ss), 1e-12f);
    if (TRANS) {
        const int nb = (int)(s >> 12), sl = (int)(s & 4095);
        for (int j = 0; j < 4; ++j)
            out[(((long long)nb * 128) + lane * 4 + j) * 4096 + sl] = (bf16_t)(vv[j] * sc);
    } else {
        for (int j = 0; j < 4; ++j)
            out[s * 128 + lane * 4 + j] = (bf16_t)(vv[j] * sc);
    }
}

// Per-position LayerNorm over C=128 (one position per wave), f32 in -> bf16 out.
__global__ __launch_bounds__(256) void ln_kernel(
    const float* __restrict__ in, const float* __restrict__ gg,
    const float* __restrict__ bb, bf16_t* __restrict__ out)
{
    const int wave = threadIdx.x >> 5, lane = threadIdx.x & 31;
    const long long s = (long long)blockIdx.x * 8 + wave;
    const float4 v = *(const float4*)(in + s * 128 + lane * 4);
    float vv[4] = { v.x, v.y, v.z, v.w };
    float sum = vv[0] + vv[1] + vv[2] + vv[3];
    float sq  = vv[0]*vv[0] + vv[1]*vv[1] + vv[2]*vv[2] + vv[3]*vv[3];
    for (int o = 16; o >= 1; o >>= 1) {
        sum += __shfl_xor(sum, o, 32);
        sq  += __shfl_xor(sq,  o, 32);
    }
    const float m = sum * (1.0f / 128.0f);
    const float rs = rsqrtf(sq * (1.0f / 128.0f) - m * m + 1e-5f);
    for (int j = 0; j < 4; ++j)
        out[s * 128 + lane * 4 + j] =
            (bf16_t)((vv[j] - m) * rs * gg[lane * 4 + j] + bb[lane * 4 + j]);
}

__global__ void cvt_kernel(const float* __restrict__ in, bf16_t* __restrict__ out, int n) {
    const int i = blockIdx.x * 256 + threadIdx.x;
    if (i < n) out[i] = (bf16_t)in[i];
}

// lstm_in[s][128+ch] = c_input (NCHW f32 -> position-major bf16). h_tm1 = input 'c'.
__global__ void gather_c_kernel(const float* __restrict__ cin, bf16_t* __restrict__ lstmIn) {
    const long long i = (long long)blockIdx.x * 256 + threadIdx.x;  // over S*128
    if (i >= 65536LL * 128) return;
    const int ch = (int)(i & 127);
    const long long s = i >> 7;
    const int nb = (int)(s >> 12), sl = (int)(s & 4095), y = sl >> 6, xx = sl & 63;
    lstmIn[s * 256 + 128 + ch] =
        (bf16_t)cin[(((long long)nb * 128 + ch) * 64 + y) * 64 + xx];
}

// Gates + state update. c_tm1 = input 'h'. d_out = [h_t | c_t] in NCHW f32.
__global__ void lstm_gates_kernel(const bf16_t* __restrict__ mix,
                                  const float* __restrict__ cprev_src,
                                  float* __restrict__ out) {
    const long long i = (long long)blockIdx.x * 256 + threadIdx.x;  // over S*128
    if (i >= 65536LL * 128) return;
    const int ch = (int)(i & 127);
    const long long s = i >> 7;
    const bf16_t* mr = mix + s * 512;
    const float fg = 1.0f / (1.0f + expf(-(float)mr[ch]));
    const float ig = 1.0f / (1.0f + expf(-(float)mr[128 + ch]));
    const float og = 1.0f / (1.0f + expf(-(float)mr[256 + ch]));
    const float cl = tanhf((float)mr[384 + ch]);
    const int nb = (int)(s >> 12), sl = (int)(s & 4095), y = sl >> 6, xx = sl & 63;
    const long long nchw = (((long long)nb * 128 + ch) * 64 + y) * 64 + xx;
    const float ct = fg * cprev_src[nchw] + ig * cl;
    out[nchw] = og * tanhf(ct);           // h_t
    out[8388608LL + nchw] = ct;           // c_t
}

// ---------------------------------------------------------------------------
extern "C" void kernel_launch(void* const* d_in, const int* in_sizes, int n_in,
                              void* d_out, int out_size, void* d_ws, size_t ws_size,
                              hipStream_t stream)
{
    (void)in_sizes; (void)n_in; (void)out_size; (void)ws_size;
    const float* x      = (const float*)d_in[0];
    const float* h_in   = (const float*)d_in[1];   // c_tm1 in reference
    const float* c_in   = (const float*)d_in[2];   // h_tm1 in reference
    const float* conv_w = (const float*)d_in[3];
    const float* conv_b = (const float*)d_in[4];
    const float* ln_g   = (const float*)d_in[5];
    const float* ln_b   = (const float*)d_in[6];
    const float* q_w = (const float*)d_in[7];  const float* q_s = (const float*)d_in[8];  const float* q_o = (const float*)d_in[9];
    const float* k_w = (const float*)d_in[10]; const float* k_s = (const float*)d_in[11]; const float* k_o = (const float*)d_in[12];
    const float* v_w = (const float*)d_in[13]; const float* v_s = (const float*)d_in[14]; const float* v_o = (const float*)d_in[15];
    const float* l_w = (const float*)d_in[16]; const float* l_s = (const float*)d_in[17]; const float* l_o = (const float*)d_in[18];
    const float* fc1_w = (const float*)d_in[19]; const float* fc1_b = (const float*)d_in[20];
    const float* fc2_w = (const float*)d_in[21]; const float* fc2_b = (const float*)d_in[22];
    const float* lstm_w = (const float*)d_in[23]; const float* lstm_b = (const float*)d_in[24];

    const long long S = 65536; // 16*64*64 positions
    char* wsb = (char*)d_ws;
    size_t off = 0;
    auto alloc = [&](size_t bytes) -> char* {
        size_t a = (off + 255) & ~(size_t)255;
        off = a + bytes;
        return wsb + a;
    };

    bf16_t* p_wconv = (bf16_t*)alloc(73728 * 2);
    bf16_t* p_wq    = (bf16_t*)alloc(16384 * 2);
    bf16_t* p_wk    = (bf16_t*)alloc(16384 * 2);
    bf16_t* p_wv    = (bf16_t*)alloc(16384 * 2);
    bf16_t* p_wl    = (bf16_t*)alloc(16384 * 2);
    bf16_t* p_wfc1  = (bf16_t*)alloc(65536 * 2);
    bf16_t* p_wfc2  = (bf16_t*)alloc(65536 * 2);
    bf16_t* p_wlstm = (bf16_t*)alloc(131072 * 2);

    float*  p_xcF  = (float*) alloc(S * 128 * 4);   // xc f32, live through l-GEMM residual
    bf16_t* p_xcBF = (bf16_t*)alloc(S * 128 * 2);   // xc bf16; reused as xb_ln bf16
    char*   qkreg  = alloc(S * 128 * 8);            // qF | kF; reused: xbF (over qF), mix bf16 (whole)
    float*  p_qF   = (float*)qkreg;
    float*  p_kF   = (float*)(qkreg + (size_t)(S * 128 * 4));
    float*  p_xbF  = p_qF;                          // after l2norm(q) done
    bf16_t* p_mix  = (bf16_t*)qkreg;                // after LN(xb) done
    bf16_t* p_qBF  = (bf16_t*)alloc(S * 128 * 2);
    bf16_t* p_kT   = (bf16_t*)alloc(S * 128 * 2);   // reused as y bf16 after f-GEMM
    bf16_t* p_yBF  = p_kT;
    bf16_t* p_vT   = (bf16_t*)alloc(S * 128 * 2);
    bf16_t* p_fT   = (bf16_t*)alloc(16 * 128 * 128 * 2);
    bf16_t* p_xbln = p_xcBF;
    bf16_t* p_mid  = (bf16_t*)alloc(S * 512 * 2);
    bf16_t* p_lin  = (bf16_t*)alloc(S * 256 * 2);   // [xu | h_tm1]

    auto cvt = [&](const float* src, bf16_t* dst, int n) {
        cvt_kernel<<<(n + 255) / 256, 256, 0, stream>>>(src, dst, n);
    };
    cvt(conv_w, p_wconv, 73728);
    cvt(q_w, p_wq, 16384); cvt(k_w, p_wk, 16384);
    cvt(v_w, p_wv, 16384); cvt(l_w, p_wl, 16384);
    cvt(fc1_w, p_wfc1, 65536); cvt(fc2_w, p_wfc2, 65536);
    cvt(lstm_w, p_wlstm, 131072);

    // conv3x3s2 + bias + LN  -> xc
    conv_ln_kernel<<<4096, 256, 0, stream>>>(x, p_wconv, conv_b, ln_g, ln_b, p_xcF, p_xcBF);

    // q,k,v 1x1 convs + affine (+relu for v, written transposed)
    gemm_bf16_wmma<1,0><<<dim3(2048,1,1), 256, 0, stream>>>(
        p_xcBF, 0, p_wq, 0, 65536, 128, 128, q_s, q_o, nullptr, p_qF, nullptr, 0, nullptr);
    gemm_bf16_wmma<1,0><<<dim3(2048,1,1), 256, 0, stream>>>(
        p_xcBF, 0, p_wk, 0, 65536, 128, 128, k_s, k_o, nullptr, p_kF, nullptr, 0, nullptr);
    gemm_bf16_wmma<2,1><<<dim3(2048,1,1), 256, 0, stream>>>(
        p_xcBF, 0, p_wv, 0, 65536, 128, 128, v_s, v_o, nullptr, nullptr, nullptr, 0, p_vT);

    // per-position L2 norms
    l2norm_kernel<0><<<8192, 256, 0, stream>>>(p_qF, p_qBF);
    l2norm_kernel<1><<<8192, 256, 0, stream>>>(p_kF, p_kT);

    // f[n] = kf^T @ vf  (M=128, K=4096, per-batch), store f^T bf16
    gemm_bf16_wmma<0,2><<<dim3(4,1,16), 256, 0, stream>>>(
        p_kT, 524288, p_vT, 524288, 128, 4096, 128,
        nullptr, nullptr, nullptr, nullptr, nullptr, 0, p_fT);

    // y = qf @ f (M=4096/batch, K=128)
    gemm_bf16_wmma<0,0><<<dim3(128,1,16), 256, 0, stream>>>(
        p_qBF, 524288, p_fT, 16384, 4096, 128, 128,
        nullptr, nullptr, nullptr, nullptr, p_yBF, 524288, nullptr);

    // Wy = relu(affine(l_w @ y)) + 2*xc  (attention residual + block residual)
    gemm_bf16_wmma<5,0><<<dim3(2048,1,1), 256, 0, stream>>>(
        p_yBF, 0, p_wl, 0, 65536, 128, 128, l_s, l_o, p_xcF, p_xbF, nullptr, 0, nullptr);

    // second LayerNorm
    ln_kernel<<<8192, 256, 0, stream>>>(p_xbF, ln_g, ln_b, p_xbln);

    // MLP: fc1 (+gelu) -> fc2 (+bias, written into lstm_in cols 0..127)
    gemm_bf16_wmma<4,0><<<dim3(2048,4,1), 256, 0, stream>>>(
        p_xbln, 0, p_wfc1, 0, 65536, 128, 512, fc1_b, nullptr, nullptr,
        nullptr, p_mid, 0, nullptr);
    gemm_bf16_wmma<3,0><<<dim3(2048,1,1), 256, 0, stream>>>(
        p_mid, 0, p_wfc2, 0, 65536, 512, 256, fc2_b, nullptr, nullptr,
        nullptr, p_lin, 0, nullptr);

    // concat h_tm1 (= input 'c') into lstm_in cols 128..255
    gather_c_kernel<<<32768, 256, 0, stream>>>(c_in, p_lin);

    // LSTM mixer: mix = lstm_w @ [xu; h_tm1] + lstm_b
    gemm_bf16_wmma<3,0><<<dim3(2048,4,1), 256, 0, stream>>>(
        p_lin, 0, p_wlstm, 0, 65536, 256, 512, lstm_b, nullptr, nullptr,
        nullptr, p_mix, 0, nullptr);

    // gates + state update -> d_out = [h_t | c_t] NCHW f32
    lstm_gates_kernel<<<32768, 256, 0, stream>>>(p_mix, h_in, (float*)d_out);
}
